// Model_29265907155643
// MI455X (gfx1250) — matrix operand order repaired and checked by reading, not compiled, not run on hardware
//
#include <hip/hip_runtime.h>
#include <hip/hip_bf16.h>

typedef __attribute__((ext_vector_type(16))) __bf16 v16bf;
typedef __attribute__((ext_vector_type(8)))  __bf16 v8bf;
typedef __attribute__((ext_vector_type(8)))  float  v8f;

union BF16x16 { v16bf v; v8bf h[2]; unsigned u[8]; };

#define NB      1024      // batch
#define MITEMS  21504     // 1024 + 1024*20 conv evaluations / gathered user rows
#define NEIGH   20

// ---------------------------------------------------------------------------
// Prep: convert weights to bf16 in WMMA fragment-native packed layout:
//   Bp[((k/16)*NCOLS + n)*16 + (k%16)] = W[n][k]
// For the conv weights, K is additionally PERMUTED so each 32-wide k-step
// covers 8 kernel rows x 4 kernel cols:  kh=(s%5)*8+2*(l>>4)+((l>>2)&1)+4*((l>>3)&1), kw=(s/5)*4+(l&3)
// (A-side uses the same permutation => GEMM result unchanged, addressing
// becomes immediate-offset only).
// ---------------------------------------------------------------------------
__global__ void prep_weights_kernel(const float* __restrict__ dense_w,   // [128,896]
                                    const float* __restrict__ item_w,    // [128,128]
                                    const float* __restrict__ user_w,    // [128,128]
                                    const float* __restrict__ last_w,    // [128,128]
                                    const float* __restrict__ conv_item_w, // [8,800]
                                    __bf16* denseBp, __bf16* itemBp,
                                    __bf16* userBp,  __bf16* lastBp,
                                    __bf16* convBp) {
  int stride = gridDim.x * blockDim.x;
  int i0 = blockIdx.x * blockDim.x + threadIdx.x;
  for (int e = i0; e < 896 * 128; e += stride) {          // 56 k-blocks x 128 cols
    int kk = e & 15, t = e >> 4, n = t & 127, kb = t >> 7;
    denseBp[e] = (__bf16)dense_w[n * 896 + kb * 16 + kk];
  }
  for (int e = i0; e < 128 * 128; e += stride) {          // 8 k-blocks x 128 cols
    int kk = e & 15, t = e >> 4, n = t & 127, kb = t >> 7;
    int src = n * 128 + kb * 16 + kk;
    itemBp[e] = (__bf16)item_w[src];
    userBp[e] = (__bf16)user_w[src];
    lastBp[e] = (__bf16)last_w[src];
  }
  for (int e = i0; e < 50 * 16 * 16; e += stride) {       // 50 half-blocks x 16 cols
    int ll = e & 15, t = e >> 4, n = t & 15, kb = t >> 4; // kb = 2*s + half
    int s = kb >> 1, half = kb & 1;
    int l = half * 16 + ll;                                // l = 0..31 within k-step
    int kh = (s % 5) * 8 + 2 * (l >> 4) + ((l >> 2) & 1) + ((l >> 3) & 1) * 4;
    int kw = (s / 5) * 4 + (l & 3);
    convBp[e] = (__bf16)conv_item_w[(n & 7) * 800 + kh * 20 + kw];
  }
}

// ---------------------------------------------------------------------------
// Shared 16x(K=128)x128 WMMA GEMM stage: Alds bf16 [16][128], packed B (global)
// ---------------------------------------------------------------------------
__device__ __forceinline__ void gemm128_stage(const __bf16 (*Alds)[128],
                                              const __bf16* __restrict__ Bpk,
                                              const float* __restrict__ bias,
                                              bool relu, float* __restrict__ outp,
                                              int row0, int tid) {
  int lane = tid & 31, wid = tid >> 5;
  int half = lane >> 4, mrow = lane & 15, ncol = lane & 15;
  v8f zero = {};
  v8f acc[2] = {zero, zero};
#pragma unroll
  for (int s = 0; s < 4; ++s) {
    int K0 = s * 32;
    BF16x16 a;
    a.h[0] = *(const v8bf*)&Alds[mrow][K0 + half * 16];        // ds_load_b128
    a.h[1] = *(const v8bf*)&Alds[mrow][K0 + half * 16 + 8];   // ds_load_b128
#pragma unroll
    for (int p = 0; p < 2; ++p) {
      int n0 = (wid * 2 + p) * 16;
      const __bf16* bp = Bpk + ((size_t)(2 * s + half) * 128 + n0 + ncol) * 16;
      BF16x16 b;
      b.h[0] = *(const v8bf*)bp;                               // global_load_b128
      b.h[1] = *(const v8bf*)(bp + 8);                         // global_load_b128
      acc[p] = __builtin_amdgcn_wmma_f32_16x16x32_bf16(false, a.v, false, b.v,
                                                       (short)0, acc[p], false, false);
    }
  }
#pragma unroll
  for (int p = 0; p < 2; ++p) {
    int n = (wid * 2 + p) * 16 + ncol;
    float bv = bias[n];
#pragma unroll
    for (int j = 0; j < 8; ++j) {
      int m = j + half * 8;
      float v = acc[p][j] + bv;
      if (relu) v = v > 0.f ? v : 0.f;
      outp[(size_t)(row0 + m) * 128 + n] = v;
    }
  }
}

// ---------------------------------------------------------------------------
// Kernel 1: per-item CNN. One block (4 waves) per item.
// conv_item implicit GEMM: C[336,8] = patches[336,800] x W[800,8] via WMMA.
// Tile t == output row oh; 16 output cols == 16 WMMA rows. With the permuted
// K ordering, every A-fragment slot is a ds_load_b32 at a compile-time
// immediate offset from one per-lane base register (bumped +832B per k-step
// within a kw-group, +8B per group); B is a linear LDS stream.
// ---------------------------------------------------------------------------
__global__ __launch_bounds__(128) void conv_kernel(
    const int* __restrict__ news_indices, const int* __restrict__ user_indices,
    const int* __restrict__ user_news,   const int* __restrict__ news_title,
    const int* __restrict__ news_entity, const int* __restrict__ news_group,
    const float* __restrict__ word_emb,  const float* __restrict__ group_emb,
    const __bf16* __restrict__ convBp,   const float* __restrict__ conv_item_b,
    const float* __restrict__ conv_title_w, const float* __restrict__ conv_title_b,
    __bf16* __restrict__ featbuf) {
  __shared__ __bf16 ig[80][52];
  __shared__ __bf16 tt[10][52];
  __shared__ __bf16 wBlds[50 * 16 * 16];     // packed conv weights, 25.6 KB
  __shared__ float convo[336][8];
  __shared__ float titleo[5][16][8];
  __shared__ int ent[40], grp[40], twd[10];

  int tid = threadIdx.x;
  int item = blockIdx.x;
  int id;
  if (item < NB) id = news_indices[item];
  else { int q = item - NB; id = user_news[user_indices[q / NEIGH] * NEIGH + (q % NEIGH)]; }

  if (tid < 40)       ent[tid]      = news_entity[id * 40 + tid];
  else if (tid < 80)  grp[tid - 40] = news_group[id * 40 + (tid - 40)];
  else if (tid < 90)  twd[tid - 80] = news_title[id * 10 + (tid - 80)];
  __syncthreads();

  // gather item/group interleaved [80,50] and title [10,50] into LDS (bf16)
  for (int e = tid; e < 4000; e += 128) {
    int r = e / 50, c = e % 50;
    float v = (r & 1) ? group_emb[grp[r >> 1] * 50 + c]
                      : word_emb[(size_t)ent[r >> 1] * 50 + c];
    ig[r][c] = (__bf16)v;
  }
  for (int e = tid; e < 500; e += 128) {
    int r = e / 50, c = e % 50;
    tt[r][c] = (__bf16)word_emb[(size_t)twd[r] * 50 + c];
  }
  {
    const uint4* s4 = (const uint4*)convBp;
    uint4* d4 = (uint4*)wBlds;
    for (int e = tid; e < 1600; e += 128) d4[e] = s4[e];     // b128 copies
  }
  __syncthreads();

  int lane = tid & 31, wid = tid >> 5;
  int half = lane >> 4, mrow = lane & 15, ncol = lane & 15;
  v8f zero = {};
  const char* igb = (const char*)&ig[0][0];

  // conv_item via WMMA: 21 row tiles distributed across 4 waves
  for (int t = wid; t < 21; t += 4) {
    // base = row (2t + 2*half) of ig, col 2*mrow  (bytes)
    const char* abase = igb + (2 * t + 2 * half) * 104 + 4 * mrow;
    const __bf16* bp = &wBlds[((unsigned)half * 16 + ncol) * 16];
    v8f acc = zero;
#pragma unroll 1
    for (int kwg = 0; kwg < 5; ++kwg) {      // kw groups of 4 columns
#pragma unroll
      for (int i = 0; i < 5; ++i) {          // kh groups of 8 rows
        BF16x16 a, b;
        const char* ab = abase + i * 832;    // 8 rows * 104 B
        a.u[0] = *(const unsigned*)(ab + 0);     // (l>>2, l&3) = (0,0)
        a.u[1] = *(const unsigned*)(ab + 4);     // (0,2)
        a.u[2] = *(const unsigned*)(ab + 104);   // (1,0)
        a.u[3] = *(const unsigned*)(ab + 108);   // (1,2)
        a.u[4] = *(const unsigned*)(ab + 416);   // (4,0)
        a.u[5] = *(const unsigned*)(ab + 420);   // (4,2)
        a.u[6] = *(const unsigned*)(ab + 520);   // (5,0)
        a.u[7] = *(const unsigned*)(ab + 524);   // (5,2)
        const __bf16* bq = bp + i * 512;
        b.h[0] = *(const v8bf*)bq;               // ds_load_b128
        b.h[1] = *(const v8bf*)(bq + 8);
        acc = __builtin_amdgcn_wmma_f32_16x16x32_bf16(false, a.v, false, b.v,
                                                      (short)0, acc, false, false);
      }
      abase += 8;                            // next 4 kw columns
      bp += 2560;                            // next 5 packed k-steps
    }

    float bias = conv_item_b[ncol & 7];
    if (ncol < 8) {
#pragma unroll
      for (int j = 0; j < 8; ++j) {
        int m = j + half * 8;                // m == ow, t == oh
        float v = acc[j] + bias;
        convo[t * 16 + m][ncol] = v > 0.f ? v : 0.f;
      }
    }
  }

  // title conv (K=40, tiny) via VALU
  for (int e = tid; e < 640; e += 128) {
    int ch = e & 7, rem = e >> 3;
    int ow = rem & 15, oh = rem >> 4;        // oh<5, ow<16
    float s = conv_title_b[ch];
    for (int kh = 0; kh < 2; ++kh)
      for (int kw = 0; kw < 20; ++kw)
        s += (float)tt[oh * 2 + kh][ow * 2 + kw] * conv_title_w[ch * 40 + kh * 20 + kw];
    titleo[oh][ow][ch] = s > 0.f ? s : 0.f;
  }
  __syncthreads();

  // maxpool + feature assembly -> featbuf[item][896] (bf16)
  __bf16* fout = featbuf + (size_t)item * 896;
  for (int f = tid; f < 896; f += 128) {
    float v;
    if (f < 640) {                            // p_item: [10,8,8] win(3,2) str(2,2)
      int ch = f & 7, w = (f >> 3) & 7, h = f >> 6;
      v = convo[(2 * h) * 16 + 2 * w][ch];
      for (int dh = 0; dh < 3; ++dh)
        for (int dw = 0; dw < 2; ++dw)
          v = fmaxf(v, convo[(2 * h + dh) * 16 + (2 * w + dw)][ch]);
    } else {                                  // p_title: [4,8,8] win(2,1) str(1,2)
      int f2 = f - 640;
      int ch = f2 & 7, w = (f2 >> 3) & 7, h = f2 >> 6;
      v = fmaxf(titleo[h][2 * w][ch], titleo[h + 1][2 * w][ch]);
    }
    fout[f] = (__bf16)v;
  }
}

// ---------------------------------------------------------------------------
// Kernel 2: fused dense (K=896) + item transform (K=128) GEMM, 16 rows/block.
// K=896 loop is 2-deep software pipelined; A strips and B pointers are linear
// in the k-step so addressing is pure pointer bumps.
// ---------------------------------------------------------------------------
__global__ __launch_bounds__(128) void dense_item_kernel(
    const __bf16* __restrict__ featbuf, const __bf16* __restrict__ denseBp,
    const float* __restrict__ dense_b,  const __bf16* __restrict__ itemBp,
    const float* __restrict__ item_b,   float* __restrict__ itemvec) {
  __shared__ __bf16 Alds[16][896];
  __shared__ __bf16 Xlds[16][128];
  int tid = threadIdx.x;
  int row0 = blockIdx.x * 16;

  {
    const uint4* src = (const uint4*)(featbuf + (size_t)row0 * 896);
    uint4* dst = (uint4*)&Alds[0][0];
    for (int e = tid; e < 16 * 112; e += 128) dst[e] = src[e];   // b128 copies
  }
  __syncthreads();

  int lane = tid & 31, wid = tid >> 5;
  int half = lane >> 4, mrow = lane & 15, ncol = lane & 15;
  v8f zero = {};
  v8f acc[2] = {zero, zero};

  const __bf16* arow  = &Alds[mrow][half * 16];      // + 32*s (+8 for hi strip)
  const __bf16* bbase = denseBp + ((size_t)half * 128 + wid * 32 + ncol) * 16;

  BF16x16 a0, a1, bA0, bB0, bA1, bB1;
  auto load_s = [&](int s, BF16x16& A, BF16x16& B0, BF16x16& B1) {
    const __bf16* ap = arow + 32 * s;
    A.h[0] = *(const v8bf*)ap;                        // ds_load_b128
    A.h[1] = *(const v8bf*)(ap + 8);
    const __bf16* bp0 = bbase + (size_t)s * 4096;     // +8KB per k-step
    B0.h[0] = *(const v8bf*)bp0;                      // global_load_b128
    B0.h[1] = *(const v8bf*)(bp0 + 8);
    const __bf16* bp1 = bp0 + 16 * 16;                // next 16-col tile
    B1.h[0] = *(const v8bf*)bp1;
    B1.h[1] = *(const v8bf*)(bp1 + 8);
    __builtin_prefetch(bp0 + 8192, 0, 1);             // 2 k-steps ahead
  };

  load_s(0, a0, bA0, bB0);
#pragma unroll 1
  for (int sp = 0; sp < 13; ++sp) {                   // steps 0..25, two per trip
    load_s(2 * sp + 1, a1, bA1, bB1);
    acc[0] = __builtin_amdgcn_wmma_f32_16x16x32_bf16(false, a0.v, false, bA0.v,
                                                     (short)0, acc[0], false, false);
    acc[1] = __builtin_amdgcn_wmma_f32_16x16x32_bf16(false, a0.v, false, bB0.v,
                                                     (short)0, acc[1], false, false);
    load_s(2 * sp + 2, a0, bA0, bB0);
    acc[0] = __builtin_amdgcn_wmma_f32_16x16x32_bf16(false, a1.v, false, bA1.v,
                                                     (short)0, acc[0], false, false);
    acc[1] = __builtin_amdgcn_wmma_f32_16x16x32_bf16(false, a1.v, false, bB1.v,
                                                     (short)0, acc[1], false, false);
  }
  load_s(27, a1, bA1, bB1);                           // steps 26, 27
  acc[0] = __builtin_amdgcn_wmma_f32_16x16x32_bf16(false, a0.v, false, bA0.v,
                                                   (short)0, acc[0], false, false);
  acc[1] = __builtin_amdgcn_wmma_f32_16x16x32_bf16(false, a0.v, false, bB0.v,
                                                   (short)0, acc[1], false, false);
  acc[0] = __builtin_amdgcn_wmma_f32_16x16x32_bf16(false, a1.v, false, bA1.v,
                                                   (short)0, acc[0], false, false);
  acc[1] = __builtin_amdgcn_wmma_f32_16x16x32_bf16(false, a1.v, false, bB1.v,
                                                   (short)0, acc[1], false, false);

#pragma unroll
  for (int p = 0; p < 2; ++p) {
    int n = (wid * 2 + p) * 16 + ncol;
    float bv = dense_b[n];
#pragma unroll
    for (int j = 0; j < 8; ++j) {
      int m = j + half * 8;
      Xlds[m][n] = (__bf16)(acc[p][j] + bv);
    }
  }
  __syncthreads();
  gemm128_stage(Xlds, itemBp, item_b, true, itemvec, row0, tid);
}

// ---------------------------------------------------------------------------
// Kernel 3: user-embedding gather + user transform GEMM (K=128)
// rows 0..1023 = user0 (self), rows 1024.. = news1 neighbors [b][m]
// ---------------------------------------------------------------------------
__global__ __launch_bounds__(128) void user_transform_kernel(
    const float* __restrict__ user_emb, const int* __restrict__ user_indices,
    const int* __restrict__ news_indices, const int* __restrict__ news_user,
    const __bf16* __restrict__ userBp, const float* __restrict__ user_b,
    float* __restrict__ uservec) {
  __shared__ __bf16 Alds[16][128];
  __shared__ int uid[16];
  int tid = threadIdx.x;
  int row0 = blockIdx.x * 16;
  if (tid < 16) {
    int r = row0 + tid;
    int u;
    if (r < NB) u = user_indices[r];
    else { int q = r - NB; u = news_user[news_indices[q / NEIGH] * NEIGH + (q % NEIGH)]; }
    uid[tid] = u;
  }
  __syncthreads();
  for (int e = tid; e < 16 * 128; e += 128) {
    int m = e >> 7, c = e & 127;
    Alds[m][c] = (__bf16)user_emb[(size_t)uid[m] * 128 + c];
  }
  __syncthreads();
  gemm128_stage(Alds, userBp, user_b, true, uservec, row0, tid);
}

// ---------------------------------------------------------------------------
// Kernel 4: capsule routing, 1 wave per (branch, sample). 7 iterations.
// branch 0 = user (self=uservec[b], neigh=itemvec[1024+b*20+m])
// branch 1 = news (self=itemvec[b], neigh=uservec[1024+b*20+m])
// Writes routed vectors to ws and straight into d_out (user_emb / news_emb).
// ---------------------------------------------------------------------------
__global__ __launch_bounds__(32) void routing_kernel(
    const float* __restrict__ uservec, const float* __restrict__ itemvec,
    float* __restrict__ routed, float* __restrict__ d_out) {
  int b2 = blockIdx.x;
  int branch = b2 >> 10, b = b2 & 1023;
  const float* self  = (branch == 0) ? uservec + (size_t)b * 128 : itemvec + (size_t)b * 128;
  const float* neigh = (branch == 0) ? itemvec + (size_t)(NB + b * NEIGH) * 128
                                     : uservec + (size_t)(NB + b * NEIGH) * 128;
  __shared__ float u0[128], u[128], z[NEIGH][128];
  __shared__ float dots[NEIGH][8], p[NEIGH][8], nrm[8];
  int t = threadIdx.x;
  const float eps = 1e-12f;

  for (int e = t; e < 128; e += 32) u0[e] = self[e];
  for (int e = t; e < NEIGH * 128; e += 32) z[e >> 7][e & 127] = neigh[e];
  __syncthreads();
  if (t < 8) {
    float s = 0.f;
    for (int d = 0; d < 16; ++d) { float x = u0[t * 16 + d]; s += x * x; }
    nrm[t] = 1.f / fmaxf(sqrtf(s), eps);
  }
  __syncthreads();
  for (int e = t; e < 128; e += 32) { u0[e] *= nrm[e >> 4]; u[e] = u0[e]; }
  for (int i = t; i < NEIGH * 8; i += 32) {
    int m = i >> 3, k = i & 7;
    float s = 0.f;
    for (int d = 0; d < 16; ++d) { float x = z[m][k * 16 + d]; s += x * x; }
    dots[m][k] = 1.f / fmaxf(sqrtf(s), eps);
  }
  __syncthreads();
  for (int e = t; e < NEIGH * 128; e += 32) {
    int m = e >> 7, c = e & 127;
    z[m][c] *= dots[m][c >> 4];
  }
  __syncthreads();

  for (int it = 0; it < 7; ++it) {
    for (int i = t; i < NEIGH * 8; i += 32) {
      int m = i >> 3, k = i & 7;
      float s = 0.f;
      for (int d = 0; d < 16; ++d) s += u[k * 16 + d] * z[m][k * 16 + d];
      dots[m][k] = s;
    }
    __syncthreads();
    if (t < NEIGH) {
      float mx = dots[t][0];
      for (int k = 1; k < 8; ++k) mx = fmaxf(mx, dots[t][k]);
      float ex[8], sum = 0.f;
      for (int k = 0; k < 8; ++k) { ex[k] = expf(dots[t][k] - mx); sum += ex[k]; }
      float inv = 1.f / sum;
      for (int k = 0; k < 8; ++k) p[t][k] = ex[k] * inv;
    }
    __syncthreads();
    for (int e = t; e < 128; e += 32) {
      int k = e >> 4;
      float val = u0[e];
      for (int m = 0; m < NEIGH; ++m) val += p[m][k] * z[m][e];
      u[e] = val;
    }
    __syncthreads();
    if (t < 8) {
      float s = 0.f;
      for (int d = 0; d < 16; ++d) { float x = u[t * 16 + d]; s += x * x; }
      nrm[t] = 1.f / fmaxf(sqrtf(s), eps);
    }
    __syncthreads();
    for (int e = t; e < 128; e += 32) u[e] *= nrm[e >> 4];
    __syncthreads();
  }
  for (int e = t; e < 128; e += 32) {
    routed[(size_t)b2 * 128 + e] = u[e];
    // d_out layout: [scores | sigmoid | label | user_emb(B*128) | news_emb(B*128)]
    d_out[3 * NB + (size_t)branch * (NB * 128) + (size_t)b * 128 + e] = u[e];
  }
}

// ---------------------------------------------------------------------------
// Kernel 5: last linear GEMM on both routed branches (M=2048, K=128, N=128)
// ---------------------------------------------------------------------------
__global__ __launch_bounds__(128) void last_linear_kernel(
    const float* __restrict__ routed, const __bf16* __restrict__ lastBp,
    const float* __restrict__ last_b, float* __restrict__ maps) {
  __shared__ __bf16 Alds[16][128];
  int tid = threadIdx.x;
  int row0 = blockIdx.x * 16;
  for (int e = tid; e < 16 * 128; e += 128) {
    int m = e >> 7, c = e & 127;
    Alds[m][c] = (__bf16)routed[(size_t)(row0 + m) * 128 + c];
  }
  __syncthreads();
  gemm128_stage(Alds, lastBp, last_b, false, maps, row0, tid);
}

// ---------------------------------------------------------------------------
// Kernel 6: scores = sum(x_map * y_map), sigmoid, label
// ---------------------------------------------------------------------------
__global__ void score_kernel(const float* __restrict__ maps, float* __restrict__ d_out) {
  int b = blockIdx.x * blockDim.x + threadIdx.x;
  if (b < NB) {
    const float* x = maps + (size_t)b * 128;          // user branch
    const float* y = maps + (size_t)(NB + b) * 128;   // news branch
    float s = 0.f;
    for (int n = 0; n < 128; ++n) s += x[n] * y[n];
    d_out[b] = s;
    d_out[NB + b] = 1.f / (1.f + expf(-s));
    d_out[2 * NB + b] = (s > 0.5f) ? 1.0f : 0.0f;
  }
}

// ---------------------------------------------------------------------------
extern "C" void kernel_launch(void* const* d_in, const int* in_sizes, int n_in,
                              void* d_out, int out_size, void* d_ws, size_t ws_size,
                              hipStream_t stream) {
  (void)in_sizes; (void)n_in; (void)out_size; (void)ws_size;
  const float* group_emb    = (const float*)d_in[0];
  const float* user_emb     = (const float*)d_in[1];
  const float* word_emb     = (const float*)d_in[2];
  const float* user_w       = (const float*)d_in[3];
  const float* user_b       = (const float*)d_in[4];
  const float* item_w       = (const float*)d_in[5];
  const float* item_b       = (const float*)d_in[6];
  const float* conv_item_w  = (const float*)d_in[7];
  const float* conv_item_b  = (const float*)d_in[8];
  const float* conv_title_w = (const float*)d_in[9];
  const float* conv_title_b = (const float*)d_in[10];
  const float* dense_w      = (const float*)d_in[11];
  const float* dense_b      = (const float*)d_in[12];
  const float* last_w       = (const float*)d_in[13];
  const float* last_b       = (const float*)d_in[14];
  const int* user_indices   = (const int*)d_in[15];
  const int* news_indices   = (const int*)d_in[16];
  const int* user_news      = (const int*)d_in[17];
  const int* news_user      = (const int*)d_in[18];
  const int* news_title     = (const int*)d_in[19];
  const int* news_entity    = (const int*)d_in[20];
  const int* news_group     = (const int*)d_in[21];
  float* out = (float*)d_out;

  char* ws = (char*)d_ws;
  size_t off = 0;
  auto carve = [&](size_t bytes) -> void* {
    void* p = ws + off;
    off = (off + bytes + 255) & ~(size_t)255;
    return p;
  };
  __bf16* denseBp = (__bf16*)carve(896 * 128 * sizeof(__bf16));
  __bf16* itemBp  = (__bf16*)carve(128 * 128 * sizeof(__bf16));
  __bf16* userBp  = (__bf16*)carve(128 * 128 * sizeof(__bf16));
  __bf16* lastBp  = (__bf16*)carve(128 * 128 * sizeof(__bf16));
  __bf16* convBp  = (__bf16*)carve(50 * 16 * 16 * sizeof(__bf16));
  __bf16* featbuf = (__bf16*)carve((size_t)MITEMS * 896 * sizeof(__bf16));
  float*  itemvec = (float*)carve((size_t)MITEMS * 128 * sizeof(float));
  float*  uservec = (float*)carve((size_t)MITEMS * 128 * sizeof(float));
  float*  routed  = (float*)carve((size_t)2 * NB * 128 * sizeof(float));
  float*  maps    = (float*)carve((size_t)2 * NB * 128 * sizeof(float));

  prep_weights_kernel<<<64, 256, 0, stream>>>(dense_w, item_w, user_w, last_w,
                                              conv_item_w, denseBp, itemBp, userBp,
                                              lastBp, convBp);
  conv_kernel<<<MITEMS, 128, 0, stream>>>(news_indices, user_indices, user_news,
                                          news_title, news_entity, news_group,
                                          word_emb, group_emb, convBp, conv_item_b,
                                          conv_title_w, conv_title_b, featbuf);
  dense_item_kernel<<<MITEMS / 16, 128, 0, stream>>>(featbuf, denseBp, dense_b,
                                                     itemBp, item_b, itemvec);
  user_transform_kernel<<<MITEMS / 16, 128, 0, stream>>>(user_emb, user_indices,
                                                         news_indices, news_user,
                                                         userBp, user_b, uservec);
  routing_kernel<<<2 * NB, 32, 0, stream>>>(uservec, itemvec, routed, out);
  last_linear_kernel<<<(2 * NB) / 16, 128, 0, stream>>>(routed, lastBp, last_b, maps);
  score_kernel<<<(NB + 127) / 128, 128, 0, stream>>>(maps, out);
}
